// PDKT_32959579030229
// MI455X (gfx1250) — compile-verified, matrix operand order checked by reading, not run on hardware
//
#include <hip/hip_runtime.h>
#include <hip/hip_bf16.h>

// ---------------------------------------------------------------------------
// PDKT model on MI455X (gfx1250, wave32, WMMA 16x16x32 f16 -> f32 acc)
//
//   Kc  cvt       : one-time f32->f16 pre-conversion of x and all weights
//   K0  init      : per-batch folded bias (bih+bhh+Wih[:,D:]@emb[c]), zero h/c
//   K1  gemm      : gates_x[t,b,4H] = x @ Wih[:, :D]^T   (WMMA, time-major out)
//   K2  lstm      : persistent 32-WG kernel, 1024 steps, h@Whh^T       (WMMA)
//   K3  gemm      : mlp_h = x @ Wm^T + bm  (f16 out)                   (WMMA)
//   K4  attn      : flash-style causal decayed softmax attention       (WMMA)
//   K5  gemm      : h1 = relu(weighted @ W1^T + b1) (f16 out)          (WMMA)
//   K6  gemm      : out = h1 @ W2^T + b2 (f32 out)                     (WMMA)
//
// All GEMM operands are pre-converted f16 -> inner loops are pure
// global_load_b128 + v_wmma (no v_cvt in the hot path).
// Workspace requirement: ~410 MB (gates_x f32 is 256 MB).
// ---------------------------------------------------------------------------

typedef __attribute__((ext_vector_type(16))) _Float16 v16h;
typedef __attribute__((ext_vector_type(8)))  float    v8f;

#define ALPHA_DK 0.6f
constexpr int Bn = 32, Ln = 1024, Dn = 512, Hn = 512, FFn = 256, OUTn = 100;
constexpr int G4 = 4 * Hn; // 2048

__device__ __forceinline__ v8f wmma32(v16h a, v16h b, v8f c) {
  return __builtin_amdgcn_wmma_f32_16x16x32_f16(false, a, false, b, (short)0, c,
                                                false, false);
}

// A/B fragment (16x32 f16): lane holds one source row; per ISA layout lanes
// 0-15 carry K {k0..k0+7, k0+16..k0+23}, lanes 16-31 carry K {+8, +24}.
// Global variant prefetches the next K window.
__device__ __forceinline__ v16h frag_rows_f16g(const _Float16* __restrict__ src,
                                               int row0, int k0, int ld, int maxrow) {
  const int lane = threadIdx.x & 31;
  int r = row0 + (lane & 15);
  if (r > maxrow) r = maxrow;
  const int kb = k0 + ((lane >> 4) << 3);
  const _Float16* p = src + (size_t)r * ld + kb;
  __builtin_prefetch(p + 32, 0, 0); // next k-step of this row
  v16h f;
#pragma unroll
  for (int i = 0; i < 8; ++i) { f[i] = p[i]; f[8 + i] = p[16 + i]; }
  return f;
}

// LDS variant (no prefetch).
__device__ __forceinline__ v16h frag_rows_f16s(const _Float16* __restrict__ src,
                                               int row0, int k0, int ld, int maxrow) {
  const int lane = threadIdx.x & 31;
  int r = row0 + (lane & 15);
  if (r > maxrow) r = maxrow;
  const int kb = k0 + ((lane >> 4) << 3);
  const _Float16* p = src + (size_t)r * ld + kb;
  v16h f;
#pragma unroll
  for (int i = 0; i < 8; ++i) { f[i] = p[i]; f[8 + i] = p[16 + i]; }
  return f;
}

// B fragment for K-major row-major source (rows = K, cols = N): lane holds one
// column (fixed N), strided gather over K rows.
__device__ __forceinline__ v16h frag_cols_f16(const _Float16* __restrict__ src,
                                              int k0, int n0, int ld) {
  const int lane = threadIdx.x & 31;
  const int n = n0 + (lane & 15);
  const int kb = k0 + ((lane >> 4) << 3);
  const _Float16* p = src + (size_t)kb * ld + n;
  v16h f;
#pragma unroll
  for (int i = 0; i < 8; ++i) { f[i] = p[(size_t)i * ld]; f[8 + i] = p[(size_t)(16 + i) * ld]; }
  return f;
}

// ---------------------------------------------------------------------------
// Kc: strided f32 -> f16 conversion (also extracts Wih[:, :D] sub-matrix).
// ---------------------------------------------------------------------------
__global__ void __launch_bounds__(256)
pdkt_cvt(const float* __restrict__ src, int sld, _Float16* __restrict__ dst,
         int dld, int rows, int cols) {
  int idx = blockIdx.x * 256 + threadIdx.x;
  const int total = rows * cols;
  if (idx < total) {
    const int r = idx / cols, c = idx - r * cols;
    dst[(size_t)r * dld + c] = (_Float16)src[(size_t)r * sld + c];
  }
}

// ---------------------------------------------------------------------------
// K0: fold concept embedding + biases into per-batch bias; zero h/c/barrier.
// ---------------------------------------------------------------------------
__global__ void __launch_bounds__(256)
pdkt_init(const int* __restrict__ concepts, const float* __restrict__ emb,
          const float* __restrict__ Wih, const float* __restrict__ bih,
          const float* __restrict__ bhh, float* __restrict__ bias_c,
          _Float16* __restrict__ hA, _Float16* __restrict__ hB,
          float* __restrict__ cst, unsigned* __restrict__ bar) {
  const int b = blockIdx.x, tid = threadIdx.x;
  __shared__ float e[Hn];
  const int c = concepts[b];
  for (int i = tid; i < Hn; i += 256) e[i] = emb[(size_t)c * Hn + i];
  for (int i = tid; i < Hn; i += 256) {
    hA[b * Hn + i] = (_Float16)0.f;
    hB[b * Hn + i] = (_Float16)0.f;
    cst[b * Hn + i] = 0.f;
  }
  if (b == 0 && tid == 0) *bar = 0u;
  __syncthreads();
  for (int j = tid; j < G4; j += 256) {
    const float* wr = Wih + (size_t)j * (2 * Dn) + Dn; // concept half of row j
    float s = bih[j] + bhh[j];
#pragma unroll 4
    for (int k = 0; k < Hn; ++k) s += wr[k] * e[k];
    bias_c[b * G4 + j] = s;
  }
}

// ---------------------------------------------------------------------------
// General C = A @ W^T (+bias)(+relu) WMMA GEMM, all-f16 operands.
// Block = 8 waves -> 128x64 tile.  tmajor: remap output row b*L+t -> t*B+b.
// ---------------------------------------------------------------------------
__global__ void __launch_bounds__(256)
pdkt_gemm_h(const _Float16* __restrict__ A, int lda,
            const _Float16* __restrict__ W, int ldw,
            const float* __restrict__ bias,
            float* __restrict__ Cf, _Float16* __restrict__ Ch, int ldc,
            int M, int N, int K, int relu, int tmajor) {
  const int lane = threadIdx.x & 31;
  const int wave = threadIdx.x >> 5;
  const int m0 = blockIdx.y * 128 + wave * 16;
  const int n0 = blockIdx.x * 64;
  const v8f vz = {0.f, 0.f, 0.f, 0.f, 0.f, 0.f, 0.f, 0.f};
  v8f acc[4] = {vz, vz, vz, vz};
  for (int k = 0; k < K; k += 32) {
    v16h a = frag_rows_f16g(A, m0, k, lda, M - 1);
#pragma unroll
    for (int nt = 0; nt < 4; ++nt) {
      v16h bf = frag_rows_f16g(W, n0 + nt * 16, k, ldw, N - 1);
      acc[nt] = wmma32(a, bf, acc[nt]);
    }
  }
  const int rb = (lane >> 4) << 3;
  const int cn = lane & 15;
#pragma unroll
  for (int nt = 0; nt < 4; ++nt) {
#pragma unroll
    for (int r = 0; r < 8; ++r) {
      const int row = m0 + rb + r;
      const int col = n0 + nt * 16 + cn;
      if (row < M && col < N) {
        float v = acc[nt][r];
        if (bias) v += bias[col];
        if (relu) v = fmaxf(v, 0.f);
        const size_t orow =
            tmajor ? ((size_t)(row & (Ln - 1)) * Bn + (row >> 10)) : (size_t)row;
        if (Cf) Cf[orow * ldc + col] = v;
        if (Ch) Ch[orow * ldc + col] = (_Float16)v;
      }
    }
  }
}

// ---------------------------------------------------------------------------
// K2: persistent LSTM.  32 blocks; block j owns hidden units [j*16, j*16+16)
// and computes all four gates for them (cell update stays block-local).
// Per step: gates(32x64) = h@Whh^T (WMMA, K=512) + gates_x[t] + bias_c.
// Double-buffered f16 h; monotonic atomic grid barrier (32 blocks always
// co-resident on MI455X).  gates_x is time-major: step t reads one contiguous
// 256 KB block (L2-friendly).
// ---------------------------------------------------------------------------
__global__ void __launch_bounds__(256)
pdkt_lstm(const float* __restrict__ gates_x, const float* __restrict__ bias_c,
          const _Float16* __restrict__ Whh_h, _Float16* __restrict__ hA,
          _Float16* __restrict__ hB, float* __restrict__ cst,
          _Float16* __restrict__ out_h, unsigned* __restrict__ bar) {
  const int j = blockIdx.x;            // unit group
  const int tid = threadIdx.x, lane = tid & 31, wave = tid >> 5;
  const int gate = wave >> 1;          // 0=i 1=f 2=g 3=o
  const int m0 = (wave & 1) * 16;      // batch half
  __shared__ float act[4 * 32 * 16];   // [gate][batch][unit]
  _Float16* const hb[2] = {hA, hB};
  const v8f vz = {0.f, 0.f, 0.f, 0.f, 0.f, 0.f, 0.f, 0.f};
  const int wrow = gate * Hn + j * 16; // Whh row base for this wave's 16 cols

  for (int t = 0; t < Ln; ++t) {
    const _Float16* hprev = hb[t & 1];
    _Float16* hnext = hb[(t + 1) & 1];
    v8f acc = vz;
    for (int ks = 0; ks < 16; ++ks) {
      v16h a  = frag_rows_f16g(hprev, m0, ks * 32, Hn, Bn - 1);
      v16h bf = frag_rows_f16g(Whh_h, wrow, ks * 32, Hn, G4 - 1);
      acc = wmma32(a, bf, acc);
    }
    const int rb = (lane >> 4) << 3;
    const int n = lane & 15;
    const int col = gate * Hn + j * 16 + n;
    const float* gx_t = gates_x + (size_t)t * Bn * G4;
#pragma unroll
    for (int r = 0; r < 8; ++r) {
      const int b = m0 + rb + r;
      float v = acc[r] + gx_t[b * G4 + col] + bias_c[b * G4 + col];
      v = (gate == 2) ? tanhf(v) : 1.0f / (1.0f + expf(-v));
      act[(gate * 32 + b) * 16 + n] = v;
    }
    __syncthreads();
    for (int idx = tid; idx < 512; idx += 256) {
      const int b = idx >> 4, nn = idx & 15, u = j * 16 + nn;
      const float ig = act[(0 * 32 + b) * 16 + nn];
      const float fg = act[(1 * 32 + b) * 16 + nn];
      const float gg = act[(2 * 32 + b) * 16 + nn];
      const float og = act[(3 * 32 + b) * 16 + nn];
      const float cn2 = fg * cst[b * Hn + u] + ig * gg;
      const float hn = og * tanhf(cn2);
      cst[b * Hn + u] = cn2;
      hnext[b * Hn + u] = (_Float16)hn;
      out_h[(size_t)(b * Ln + t) * Hn + u] = (_Float16)hn;
    }
    __syncthreads();
    if (tid == 0) {
      __threadfence();
      atomicAdd(bar, 1u);
      const unsigned target = (unsigned)(t + 1) * 32u;
      while (atomicAdd(bar, 0u) < target) __builtin_amdgcn_s_sleep(2);
      __threadfence();
    }
    __syncthreads();
  }
}

// ---------------------------------------------------------------------------
// K4: flash-style causal decayed attention.  Block = (b, 16-row t tile).
// z = (Q.K^T) * exp(-a*(t-s)) masked; streaming softmax; acc += P@V (WMMA).
// ---------------------------------------------------------------------------
__global__ void __launch_bounds__(256)
pdkt_attn(const _Float16* __restrict__ mlp_h, const _Float16* __restrict__ out_h,
          _Float16* __restrict__ weighted) {
  const int bb = blockIdx.x >> 6;
  const int t0 = (blockIdx.x & 63) << 4;
  const int tid = threadIdx.x, lane = tid & 31, wave = tid >> 5;
  __shared__ _Float16 qbuf[16 * Hn];   // Q tile (16 KB)
  __shared__ float    zbuf[16 * 32];   // decayed scores for one s-chunk
  __shared__ _Float16 pbuf[16 * 32];   // probabilities (f16) for P@V
  __shared__ float mrow[16], lrow[16], cfrow[16];

  const _Float16* Q = mlp_h + (size_t)(bb * Ln + t0) * Hn;
  for (int i = tid; i < 16 * Hn; i += 256) qbuf[i] = Q[i];
  if (tid < 16) { mrow[tid] = -1e30f; lrow[tid] = 0.f; }
  __syncthreads();

  const _Float16* Kb = out_h + (size_t)bb * Ln * Hn;
  const v8f vz = {0.f, 0.f, 0.f, 0.f, 0.f, 0.f, 0.f, 0.f};
  v8f acc[4] = {vz, vz, vz, vz};       // this wave owns output cols wave*64..+63
  const int rb = (lane >> 4) << 3;
  const int cn = lane & 15;

  const int nchunks = ((t0 + 15) >> 5) + 1;
  for (int ch = 0; ch < nchunks; ++ch) {
    const int s0 = ch * 32;
    // Phase A: waves 0,1 produce the two 16x16 decayed score tiles
    if (wave < 2) {
      v8f sc = vz;
      for (int ks = 0; ks < 16; ++ks) {
        v16h a  = frag_rows_f16s(qbuf, 0, ks * 32, Hn, 15);
        v16h bf = frag_rows_f16g(Kb, s0 + wave * 16, ks * 32, Hn, Ln - 1);
        sc = wmma32(a, bf, sc);
      }
      const int scol = s0 + wave * 16 + cn;
#pragma unroll
      for (int r = 0; r < 8; ++r) {
        const int trow = t0 + rb + r;
        const float z = (scol <= trow)
                            ? sc[r] * expf(-ALPHA_DK * (float)(trow - scol))
                            : -1e30f;
        zbuf[(rb + r) * 32 + wave * 16 + cn] = z;
      }
    }
    __syncthreads();
    // Phase B: per-row streaming max/sum update, probs -> f16
    if (tid < 16) {
      const float m = mrow[tid];
      float mnew = m;
      for (int c2 = 0; c2 < 32; ++c2) mnew = fmaxf(mnew, zbuf[tid * 32 + c2]);
      const float cf = expf(m - mnew);
      float s = 0.f;
      for (int c2 = 0; c2 < 32; ++c2) {
        const float p = expf(zbuf[tid * 32 + c2] - mnew);
        pbuf[tid * 32 + c2] = (_Float16)p;
        s += p;
      }
      lrow[tid] = lrow[tid] * cf + s;
      mrow[tid] = mnew;
      cfrow[tid] = cf;
    }
    __syncthreads();
    // Phase C: rescale running acc, acc += P(16x32) @ V(32x512) via WMMA
    {
      v16h pa = frag_rows_f16s(pbuf, 0, 0, 32, 15);
#pragma unroll
      for (int nt = 0; nt < 4; ++nt) {
#pragma unroll
        for (int r = 0; r < 8; ++r) acc[nt][r] *= cfrow[rb + r];
        v16h vb = frag_cols_f16(Kb + (size_t)s0 * Hn, 0, wave * 64 + nt * 16, Hn);
        acc[nt] = wmma32(pa, vb, acc[nt]);
      }
    }
    __syncthreads();
  }
  // finalize: divide by running denominator, store f16 (feeds W1 WMMA)
#pragma unroll
  for (int nt = 0; nt < 4; ++nt)
#pragma unroll
    for (int r = 0; r < 8; ++r) {
      const int row = rb + r;
      weighted[(size_t)(bb * Ln + t0 + row) * Hn + wave * 64 + nt * 16 + cn] =
          (_Float16)(acc[nt][r] / lrow[row]);
    }
}

// ---------------------------------------------------------------------------
extern "C" void kernel_launch(void* const* d_in, const int* in_sizes, int n_in,
                              void* d_out, int out_size, void* d_ws, size_t ws_size,
                              hipStream_t stream) {
  (void)in_sizes; (void)n_in; (void)out_size; (void)ws_size;
  const float* x        = (const float*)d_in[0];
  const int*   concepts = (const int*)d_in[1];
  const float* emb      = (const float*)d_in[2];
  const float* Wih      = (const float*)d_in[3];
  const float* Whh      = (const float*)d_in[4];
  const float* bih      = (const float*)d_in[5];
  const float* bhh      = (const float*)d_in[6];
  const float* Wm       = (const float*)d_in[7];
  const float* bm       = (const float*)d_in[8];
  const float* W1       = (const float*)d_in[9];
  const float* b1       = (const float*)d_in[10];
  const float* W2       = (const float*)d_in[11];
  const float* b2       = (const float*)d_in[12];
  float* out = (float*)d_out;

  char* base = (char*)d_ws;
  size_t off = 0;
  auto carve = [&](size_t bytes) -> char* {
    char* p = base + off;
    off = (off + bytes + 255) & ~(size_t)255;
    return p;
  };
  float*    gates_x = (float*)carve((size_t)Bn * Ln * G4 * 4);      // 256 MB
  float*    bias_c  = (float*)carve((size_t)Bn * G4 * 4);
  _Float16* hA      = (_Float16*)carve((size_t)Bn * Hn * 2);
  _Float16* hB      = (_Float16*)carve((size_t)Bn * Hn * 2);
  float*    cst     = (float*)carve((size_t)Bn * Hn * 4);
  _Float16* out_h   = (_Float16*)carve((size_t)Bn * Ln * Hn * 2);   // 32 MB
  _Float16* mlp_h   = (_Float16*)carve((size_t)Bn * Ln * Dn * 2);   // 32 MB
  _Float16* wtd_h   = (_Float16*)carve((size_t)Bn * Ln * Hn * 2);   // 32 MB
  _Float16* h1_h    = (_Float16*)carve((size_t)Bn * Ln * FFn * 2);  // 16 MB
  _Float16* x_h     = (_Float16*)carve((size_t)Bn * Ln * Dn * 2);   // 32 MB
  _Float16* WihX_h  = (_Float16*)carve((size_t)G4 * Dn * 2);        // 2 MB
  _Float16* Whh_h   = (_Float16*)carve((size_t)G4 * Hn * 2);        // 2 MB
  _Float16* Wm_h    = (_Float16*)carve((size_t)Dn * Dn * 2);
  _Float16* W1_h    = (_Float16*)carve((size_t)FFn * Hn * 2);
  _Float16* W2_h    = (_Float16*)carve((size_t)OUTn * FFn * 2);
  unsigned* bar     = (unsigned*)carve(256);

  const int M = Bn * Ln; // 32768 rows

  // Kc: one-time f32->f16 conversions (also strips Wih to its x-half)
  auto cvt = [&](const float* s, int sld, _Float16* d, int dld, int r, int c) {
    const int n = r * c;
    pdkt_cvt<<<(n + 255) / 256, 256, 0, stream>>>(s, sld, d, dld, r, c);
  };
  cvt(x, Dn, x_h, Dn, M, Dn);
  cvt(Wih, 2 * Dn, WihX_h, Dn, G4, Dn);
  cvt(Whh, Hn, Whh_h, Hn, G4, Hn);
  cvt(Wm, Dn, Wm_h, Dn, Dn, Dn);
  cvt(W1, Hn, W1_h, Hn, FFn, Hn);
  cvt(W2, FFn, W2_h, FFn, OUTn, FFn);

  // K0: folded bias + state/barrier init
  pdkt_init<<<Bn, 256, 0, stream>>>(concepts, emb, Wih, bih, bhh, bias_c, hA,
                                    hB, cst, bar);
  // K1: gates_x[t,b,:] = x @ Wih[:, :D]^T   (time-major for the recurrence)
  pdkt_gemm_h<<<dim3(G4 / 64, M / 128), 256, 0, stream>>>(
      x_h, Dn, WihX_h, Dn, nullptr, gates_x, nullptr, G4, M, G4, Dn, 0, 1);
  // K2: persistent LSTM recurrence
  pdkt_lstm<<<32, 256, 0, stream>>>(gates_x, bias_c, Whh_h, hA, hB, cst, out_h,
                                    bar);
  // K3: mlp_h = (x @ Wm^T + bm) as f16
  pdkt_gemm_h<<<dim3(Dn / 64, M / 128), 256, 0, stream>>>(
      x_h, Dn, Wm_h, Dn, bm, nullptr, mlp_h, Dn, M, Dn, Dn, 0, 0);
  // K4: causal decayed attention
  pdkt_attn<<<Bn * (Ln / 16), 256, 0, stream>>>(mlp_h, out_h, wtd_h);
  // K5: h1 = relu(weighted @ W1^T + b1) as f16
  pdkt_gemm_h<<<dim3(FFn / 64, M / 128), 256, 0, stream>>>(
      wtd_h, Hn, W1_h, Hn, b1, nullptr, h1_h, FFn, M, FFn, Hn, 1, 0);
  // K6: out = h1 @ W2^T + b2 (f32)
  pdkt_gemm_h<<<dim3((OUTn + 63) / 64, M / 128), 256, 0, stream>>>(
      h1_h, FFn, W2_h, FFn, b2, out, nullptr, OUTn, M, OUTn, FFn, 0, 0);
}
